// MultiHeadAttn_91010357002583
// MI455X (gfx1250) — matrix operand order repaired and checked by reading, not compiled, not run on hardware
//
#include <hip/hip_runtime.h>

typedef __bf16 bf16;
typedef bf16  v8bf  __attribute__((ext_vector_type(8)));
typedef bf16  v16bf __attribute__((ext_vector_type(16)));
typedef float v8f   __attribute__((ext_vector_type(8)));

#define WMMA_BF16(a, b, c) \
    __builtin_amdgcn_wmma_f32_16x16x32_bf16(false, (a), false, (b), (short)0, (c), false, false)

// Problem constants
#define BB 4
#define SS 2048
#define DD 1024
#define HH 16
#define HD 64
#define ROWS (BB * SS)         // 8192
#define INV_SCALE 0.125f       // 1/sqrt(64)

// ---------------- fragment loaders (contiguous k order, wave32) ----------------

// A fragment 16x32 (bf16). p points at element (0,0) of the 16x32 tile, ld in elems.
// lane L: m = L%16, half = L/16; elem e -> k = half*16+e (contiguous)
__device__ __forceinline__ v16bf load_a_frag(const bf16* p, int ld) {
    int lane = threadIdx.x & 31;
    int m = lane & 15, half = lane >> 4;
    const bf16* row = p + m * ld + half * 16;
    union { v16bf v; struct { v8bf lo, hi; } s; } u;
    u.s.lo = *(const v8bf*)(row);
    u.s.hi = *(const v8bf*)(row + 8);
    return u.v;
}

// B fragment 32x16 (bf16) read from a row-major [N][K] ("B-transposed") buffer.
// p points at (n0, k0); lane L: n = L%16, khalf = L/16; elem e -> k = khalf*16+e (contiguous).
__device__ __forceinline__ v16bf load_bT_frag(const bf16* p, int ld) {
    int lane = threadIdx.x & 31;
    int n = lane & 15, kh = lane >> 4;
    return *(const v16bf*)(p + n * ld + kh * 16);
}

// ---------------- conversion kernels ----------------

__global__ void cvt_kernel(const float* __restrict__ src, bf16* __restrict__ dst, int n) {
    int i = blockIdx.x * blockDim.x + threadIdx.x;
    if (i < n) dst[i] = (bf16)src[i];
}

// src: row-major [K][N] f32 -> dst: row-major [N][K] bf16 (coalesced writes)
__global__ void cvtT_kernel(const float* __restrict__ src, bf16* __restrict__ dst,
                            int K, int N) {
    int i = blockIdx.x * blockDim.x + threadIdx.x;
    if (i < K * N) {
        int n = i / K, k = i - n * K;
        dst[i] = (bf16)src[k * N + n];
    }
}

// ---------------- common 32x64 WMMA tile, ping-pong double buffering ----------------

__device__ __forceinline__ void load_set(const bf16* __restrict__ a0row,
                                         const bf16* __restrict__ a1row,
                                         const bf16* __restrict__ brow, int k0,
                                         v16bf& a0, v16bf& a1, v16bf b[4]) {
    a0 = load_a_frag(a0row + k0, DD);
    a1 = load_a_frag(a1row + k0, DD);
#pragma unroll
    for (int j = 0; j < 4; ++j)
        b[j] = load_bT_frag(brow + (size_t)j * 16 * DD + k0, DD);
}

__device__ __forceinline__ void mma_set(const v16bf& a0, const v16bf& a1,
                                        const v16bf b[4], v8f c[2][4]) {
#pragma unroll
    for (int j = 0; j < 4; ++j) {
        c[0][j] = WMMA_BF16(a0, b[j], c[0][j]);
        c[1][j] = WMMA_BF16(a1, b[j], c[1][j]);
    }
}

// a0row/a1row: the two 16-row groups of A (row-major [*, DD]).
// brow: row (ntile*64) of the [N][DD] B-transposed matrix. Accumulates into c[2][4].
__device__ __forceinline__ void gemm_tile_32x64(const bf16* __restrict__ a0row,
                                                const bf16* __restrict__ a1row,
                                                const bf16* __restrict__ brow,
                                                v8f c[2][4]) {
    v16bf a0[2], a1[2], b[2][4];
    load_set(a0row, a1row, brow, 0, a0[0], a1[0], b[0]);
#pragma unroll 1
    for (int k0 = 0; k0 < DD - 64; k0 += 64) {
        load_set(a0row, a1row, brow, k0 + 32, a0[1], a1[1], b[1]);
        mma_set(a0[0], a1[0], b[0], c);
        load_set(a0row, a1row, brow, k0 + 64, a0[0], a1[0], b[0]);
        mma_set(a0[1], a1[1], b[1], c);
    }
    // tail: buf0 holds k = DD-64
    load_set(a0row, a1row, brow, DD - 32, a0[1], a1[1], b[1]);
    mma_set(a0[0], a1[0], b[0], c);
    mma_set(a0[1], a1[1], b[1], c);
}

// ---------------- GEMM1: qkv = x @ Wqkv, scatter into Q/K/Vt ----------------
// One wave per 32x64 output tile. M=8192, N=3072, K=1024.
__global__ __launch_bounds__(256) void gemm_qkv_kernel(
        const bf16* __restrict__ xb,   // [8192][1024]
        const bf16* __restrict__ wT,   // [3072][1024]  (W_qkv transposed)
        bf16* __restrict__ Q,          // [64(bh)][2048][64]
        bf16* __restrict__ Kb,         // [64(bh)][2048][64]
        bf16* __restrict__ Vt)         // [64(bh)][64(hd)][2048]
{
    int wave = (blockIdx.x * blockDim.x + threadIdx.x) >> 5;
    const int NT = 3072 / 64;                  // 48 tiles along N
    int mt = wave / NT, nt = wave % NT;
    v8f c[2][4] = {};
    gemm_tile_32x64(xb + (size_t)(mt * 32) * DD,
                    xb + (size_t)(mt * 32 + 16) * DD,
                    wT + (size_t)(nt * 64) * DD, c);

    // Epilogue scatter. 64-col tile stays within one of q/k/v and one head.
    int lane = threadIdx.x & 31;
    int n_in = lane & 15, rh = lane >> 4;
    int which = (nt * 64) / DD;                // 0=q 1=k 2=v
    int dbase = (nt * 64) % DD;
#pragma unroll
    for (int g = 0; g < 2; ++g) {
#pragma unroll
        for (int j = 0; j < 4; ++j) {
            int d = dbase + j * 16 + n_in;
            int h = d >> 6, hd = d & 63;
#pragma unroll
            for (int r = 0; r < 8; ++r) {
                int m = mt * 32 + g * 16 + rh * 8 + r;
                int b = m >> 11, s = m & 2047;
                int bh = b * HH + h;
                bf16 val = (bf16)c[g][j][r];
                if (which == 0)      Q [((size_t)bh * SS + s) * HD + hd] = val;
                else if (which == 1) Kb[((size_t)bh * SS + s) * HD + hd] = val;
                else                 Vt[((size_t)bh * HD + hd) * SS + s] = val;
            }
        }
    }
}

// ---------------- Flash attention: one wave per 16-row q-tile ----------------
__global__ __launch_bounds__(128) void attn_kernel(
        const bf16* __restrict__ Q,            // [bh][2048][64]
        const bf16* __restrict__ Kb,           // [bh][2048][64]
        const bf16* __restrict__ Vt,           // [bh][64][2048]
        const long long* __restrict__ mask,    // [B][S] int64
        bf16* __restrict__ AO)                 // [8192][1024]
{
    __shared__ bf16 pLds[4][16 * 32];          // per-wave P bounce buffer
    int wave_in = threadIdx.x >> 5;
    int wave = blockIdx.x * 4 + wave_in;
    int bh = wave >> 7;                        // / 128 q-tiles
    int qt = wave & 127;
    int b = bh >> 4, h = bh & 15;
    int s0 = qt * 16;
    int lane = threadIdx.x & 31;
    int n_in = lane & 15, rh = lane >> 4;

    const bf16* Qp = Q  + ((size_t)bh * SS + s0) * HD;
    const bf16* Kp = Kb + (size_t)bh * SS * HD;
    const bf16* Vp = Vt + (size_t)bh * HD * SS;

    v16bf aQ0 = load_a_frag(Qp,      HD);      // d = 0..31
    v16bf aQ1 = load_a_frag(Qp + 32, HD);      // d = 32..63

    v8f o0 = {}, o1 = {}, o2 = {}, o3 = {};
    float mrow[8], lrow[8];
#pragma unroll
    for (int r = 0; r < 8; ++r) { mrow[r] = -__builtin_inff(); lrow[r] = 0.f; }
    bf16* pw = pLds[wave_in];

#pragma unroll 1
    for (int kb = 0; kb < SS / 32; ++kb) {
        int n0 = kb * 32;
        // ---- scores: two 16x16 f32 tiles over HD=64 (2 WMMAs each) ----
        v8f t0 = {}, t1 = {};
        const bf16* kp0 = Kp + (size_t)n0 * HD;
        const bf16* kp1 = Kp + (size_t)(n0 + 16) * HD;
        t0 = WMMA_BF16(aQ0, load_bT_frag(kp0,      HD), t0);
        t0 = WMMA_BF16(aQ1, load_bT_frag(kp0 + 32, HD), t0);
        t1 = WMMA_BF16(aQ0, load_bT_frag(kp1,      HD), t1);
        t1 = WMMA_BF16(aQ1, load_bT_frag(kp1 + 32, HD), t1);

        // ---- hoisted V fragment loads: overlap with the softmax VALU work ----
        const bf16* vp = Vp + n0;
        v16bf vb0 = load_bT_frag(vp + (size_t) 0 * SS, SS);
        v16bf vb1 = load_bT_frag(vp + (size_t)16 * SS, SS);
        v16bf vb2 = load_bT_frag(vp + (size_t)32 * SS, SS);
        v16bf vb3 = load_bT_frag(vp + (size_t)48 * SS, SS);

        // scale + mask (mask is all-ones in reference; kept for fidelity)
        float mb0 = (mask[(size_t)b * SS + n0 + n_in]      == 0) ? -__builtin_inff() : 0.f;
        float mb1 = (mask[(size_t)b * SS + n0 + 16 + n_in] == 0) ? -__builtin_inff() : 0.f;
#pragma unroll
        for (int r = 0; r < 8; ++r) {
            t0[r] = t0[r] * INV_SCALE + mb0;
            t1[r] = t1[r] * INV_SCALE + mb1;
        }
        // ---- online softmax ----
        float mnew[8], alpha[8];
#pragma unroll
        for (int r = 0; r < 8; ++r) {
            float t = fmaxf(t0[r], t1[r]);
#pragma unroll
            for (int off = 1; off < 16; off <<= 1)
                t = fmaxf(t, __shfl_xor(t, off, 32));   // stays within 16-lane half
            mnew[r]  = fmaxf(mrow[r], t);
            alpha[r] = __expf(mrow[r] - mnew[r]);        // exp(-inf - x) = 0 on first block
            mrow[r]  = mnew[r];
        }
#pragma unroll
        for (int r = 0; r < 8; ++r) {
            t0[r] = __expf(t0[r] - mnew[r]);
            t1[r] = __expf(t1[r] - mnew[r]);
            float rs = t0[r] + t1[r];
#pragma unroll
            for (int off = 1; off < 16; off <<= 1)
                rs += __shfl_xor(rs, off, 32);
            lrow[r] = lrow[r] * alpha[r] + rs;
            o0[r] *= alpha[r]; o1[r] *= alpha[r]; o2[r] *= alpha[r]; o3[r] *= alpha[r];
        }
        // ---- P: C-layout -> A-layout via LDS ----
#pragma unroll
        for (int r = 0; r < 8; ++r) {
            int m = rh * 8 + r;
            pw[m * 32 + n_in]      = (bf16)t0[r];
            pw[m * 32 + 16 + n_in] = (bf16)t1[r];
        }
        asm volatile("s_wait_dscnt 0" ::: "memory");
        v16bf pa = load_a_frag(pw, 32);                 // P 16x32 as A fragment
        // ---- PV ----
        o0 = WMMA_BF16(pa, vb0, o0);
        o1 = WMMA_BF16(pa, vb1, o1);
        o2 = WMMA_BF16(pa, vb2, o2);
        o3 = WMMA_BF16(pa, vb3, o3);
    }
    // ---- normalize + store bf16 attn output [b,s,h*64+hd] ----
#pragma unroll
    for (int r = 0; r < 8; ++r) {
        float inv = 1.0f / lrow[r];
        int m = rh * 8 + r;
        bf16* dst = AO + ((size_t)(b * SS + s0 + m)) * DD + h * HD;
        dst[0 * 16 + n_in] = (bf16)(o0[r] * inv);
        dst[1 * 16 + n_in] = (bf16)(o1[r] * inv);
        dst[2 * 16 + n_in] = (bf16)(o2[r] * inv);
        dst[3 * 16 + n_in] = (bf16)(o3[r] * inv);
    }
}

// ---------------- GEMM2: out = AO @ Wfc + b_fc (f32 out) ----------------
__global__ __launch_bounds__(256) void gemm_fc_kernel(
        const bf16* __restrict__ AO,   // [8192][1024]
        const bf16* __restrict__ wT,   // [1024][1024] (W_fc transposed)
        const float* __restrict__ bias,
        float* __restrict__ out)       // [8192][1024]
{
    int wave = (blockIdx.x * blockDim.x + threadIdx.x) >> 5;
    const int NT = DD / 64;                    // 16 tiles along N
    int mt = wave / NT, nt = wave % NT;
    v8f c[2][4] = {};
    gemm_tile_32x64(AO + (size_t)(mt * 32) * DD,
                    AO + (size_t)(mt * 32 + 16) * DD,
                    wT + (size_t)(nt * 64) * DD, c);

    int lane = threadIdx.x & 31;
    int n_in = lane & 15, rh = lane >> 4;
#pragma unroll
    for (int g = 0; g < 2; ++g) {
#pragma unroll
        for (int j = 0; j < 4; ++j) {
            int col = nt * 64 + j * 16 + n_in;
            float bv = bias[col];
#pragma unroll
            for (int r = 0; r < 8; ++r) {
                int m = mt * 32 + g * 16 + rh * 8 + r;
                out[(size_t)m * DD + col] = c[g][j][r] + bv;
            }
        }
    }
}

// ---------------- host launcher ----------------
extern "C" void kernel_launch(void* const* d_in, const int* in_sizes, int n_in,
                              void* d_out, int out_size, void* d_ws, size_t ws_size,
                              hipStream_t stream) {
    const float*      x    = (const float*)d_in[0];
    // d_in[1] char_ids, d_in[2] seq_len: unused by the reference output
    const long long*  mask = (const long long*)d_in[3];
    const float*      Wqkv = (const float*)d_in[4];
    const float*      Wfc  = (const float*)d_in[5];
    const float*      bfc  = (const float*)d_in[6];
    float*            out  = (float*)d_out;

    char* ws = (char*)d_ws;
    size_t off = 0;
    auto alloc = [&](size_t bytes) {
        void* p = ws + off;
        off += (bytes + 255) & ~(size_t)255;
        return p;
    };
    bf16* xb  = (bf16*)alloc((size_t)ROWS * DD * 2);          // x in bf16
    bf16* wqT = (bf16*)alloc((size_t)3 * DD * DD * 2);        // Wqkv^T
    bf16* wfT = (bf16*)alloc((size_t)DD * DD * 2);            // Wfc^T
    bf16* Q   = (bf16*)alloc((size_t)BB * HH * SS * HD * 2);
    bf16* K   = (bf16*)alloc((size_t)BB * HH * SS * HD * 2);
    bf16* Vt  = (bf16*)alloc((size_t)BB * HH * HD * SS * 2);
    bf16* AO  = (bf16*)alloc((size_t)ROWS * DD * 2);

    int nX = ROWS * DD;                     // 8388608
    cvt_kernel<<<(nX + 255) / 256, 256, 0, stream>>>(x, xb, nX);
    cvtT_kernel<<<(3 * DD * DD + 255) / 256, 256, 0, stream>>>(Wqkv, wqT, DD, 3 * DD);
    cvtT_kernel<<<(DD * DD + 255) / 256, 256, 0, stream>>>(Wfc, wfT, DD, DD);

    // (8192/32) * (3072/64) = 12288 waves -> 1536 blocks of 8 waves
    gemm_qkv_kernel<<<1536, 256, 0, stream>>>(xb, wqT, Q, K, Vt);
    // 64 bh * 128 q-tiles = 8192 waves -> 2048 blocks of 4 waves
    attn_kernel<<<2048, 128, 0, stream>>>(Q, K, Vt, mask, AO);
    // (8192/32) * (1024/64) = 4096 waves -> 512 blocks of 8 waves
    gemm_fc_kernel<<<512, 256, 0, stream>>>(AO, wfT, bfc, out);
}
